// DynamicUpsamplePAC_68496138437304
// MI455X (gfx1250) — matrix operand order repaired and checked by reading, not compiled, not run on hardware
//
#include <hip/hip_runtime.h>

typedef __attribute__((ext_vector_type(16))) _Float16 v16h;
typedef __attribute__((ext_vector_type(8)))  _Float16 v8h;
typedef __attribute__((ext_vector_type(8)))  float    v8f;
typedef __attribute__((ext_vector_type(4)))  float    v4f;

#define B_    8
#define CIN   128
#define COUT  64
#define KK    5
#define HIN   160
#define TGT   256
#define HSUB  128    // sub-grid (TGT/2) and downsampled size
#define XDP   130    // padded sub-grid dim (1-pixel zero border)

// -------- workspace layout (bytes) --------
#define XDPAD_BYTES ((size_t)B_ * XDP * XDP * CIN * 2)            // 34,611,200
#define KERN_BYTES  ((size_t)B_ * 25 * TGT * TGT * 4)             // 52,428,800
#define WFRAG_ELEMS ((size_t)25 * 4 * 4 * 32 * 16)                // 204,800
#define WFRAG_BYTES (WFRAG_ELEMS * 2)

#define TAP_HALVES  (4 * 4 * 32 * 16)                             // 8192 halves = 16KB per tap
#define MAX_TAPS    9
#define LDS_BYTES   (MAX_TAPS * TAP_HALVES * 2)                   // 147,456 B dynamic LDS

// ============ Stage 1: bilinear downsample 160->128, f32 -> f16, layout (b, hs+1, ws+1, c) padded ============
__global__ void __launch_bounds__(256)
k_downsample(const float* __restrict__ x, _Float16* __restrict__ xdpad) {
  int t = blockIdx.x * 256 + threadIdx.x;     // exact: 8*128*128*128 threads
  int c  = t & 127;
  int ws = (t >> 7)  & 127;
  int hs = (t >> 14) & 127;
  int b  = t >> 21;

  float sh = fmaxf((hs + 0.5f) * 1.25f - 0.5f, 0.0f);
  int h0 = (int)sh; float fh = sh - (float)h0; int h1 = min(h0 + 1, HIN - 1);
  float sw = fmaxf((ws + 0.5f) * 1.25f - 0.5f, 0.0f);
  int w0 = (int)sw; float fw = sw - (float)w0; int w1 = min(w0 + 1, HIN - 1);

  const float* xb = x + (size_t)(b * CIN + c) * HIN * HIN;
  float v00 = xb[h0 * HIN + w0], v01 = xb[h0 * HIN + w1];
  float v10 = xb[h1 * HIN + w0], v11 = xb[h1 * HIN + w1];
  float v = (1.0f - fh) * ((1.0f - fw) * v00 + fw * v01)
          +         fh  * ((1.0f - fw) * v10 + fw * v11);

  xdpad[((size_t)(b * XDP + hs + 1) * XDP + (ws + 1)) * CIN + c] = (_Float16)v;
}

// ============ Stage 2: PAC gaussian kernel, stored parity-deinterleaved in w ============
// kern2[b][tap][h][ (w&1)*128 + (w>>1) ]
__global__ void __launch_bounds__(256)
k_pack(const float* __restrict__ guide, float* __restrict__ kern2) {
  int t = blockIdx.x * 256 + threadIdx.x;     // exact: 8*25*256*256 threads
  int w  = t & 255;
  int h  = (t >> 8) & 255;
  int tb = t >> 16;                            // b*25 + tap
  int tap = tb % 25;
  int b   = tb / 25;
  int i = tap / 5, j = tap % 5;
  int hn = h + i - 2, wn = w + j - 2;
  bool inb = (hn >= 0) && (hn < TGT) && (wn >= 0) && (wn < TGT);

  const float* g = guide + (size_t)b * 3 * TGT * TGT;
  float s = 0.0f;
#pragma unroll
  for (int ch = 0; ch < 3; ++ch) {
    float ctr = g[ch * TGT * TGT + h * TGT + w];
    float nb  = inb ? g[ch * TGT * TGT + hn * TGT + wn] : 0.0f;
    float d = nb - ctr;
    s += d * d;
  }
  float kv = __expf(-0.5f * s);
  kern2[((size_t)tb * TGT + h) * TGT + ((w & 1) * HSUB + (w >> 1))] = kv;
}

// ============ Stage 3: weight -> lane-swizzled f16 B fragments ============
// layout: wfrag[(((tap*4 + ks)*4 + nt)*32 + lane)*16 + e]
// B (32x16) 16-bit layout: lane holds column N = lane&15; elements e=0..15 give
// K = ks*32 + (lane>=16 ? 8 : 0) + (e&7) + (e>=8 ? 16 : 0)   (table order, matches A)
__global__ void __launch_bounds__(256)
k_wfrag(const float* __restrict__ weight, _Float16* __restrict__ wfrag) {
  int t = blockIdx.x * 256 + threadIdx.x;     // exact: 204,800 threads
  int e    = t & 15;
  int lane = (t >> 4) & 31;
  int nt   = (t >> 9) & 3;
  int ks   = (t >> 11) & 3;
  int tap  = t >> 13;
  int c = ks * 32 + ((lane >= 16) ? 8 : 0) + (e & 7) + ((e >> 3) << 4);
  int o = nt * 16 + (lane & 15);
  int i = tap / 5, j = tap % 5;
  // weight layout: (CIN, COUT, 5, 5)
  wfrag[t] = (_Float16)weight[(((size_t)c * COUT + o) * KK + i) * KK + j];
}

// ============ Stage 4: PAC transposed conv via per-tap WMMA GEMMs ============
// grid: b(8) x parity(4) x hs(128) = 4096 blocks, 256 threads (8 waves)
// wave: M = 16 pixels (one w-segment of the parity sub-grid row), N = 64 Cout (4 tiles), K = 128 (4 steps)
// B fragments for this parity class's taps are staged in LDS once per block.
__global__ void __launch_bounds__(256)
k_pac_wmma(const _Float16* __restrict__ xdpad,
           const float*    __restrict__ kern2,
           const _Float16* __restrict__ wfrag,
           const float*    __restrict__ bias,
           float*          __restrict__ out) {
  extern __shared__ _Float16 lds_w[];

  const int lane = threadIdx.x & 31;
  const int wave = threadIdx.x >> 5;
  const int bp   = blockIdx.x;
  const int hs   = bp & 127;
  const int par  = (bp >> 7) & 3;
  const int b    = bp >> 9;
  const int ph = par >> 1, pw = par & 1;
  const int h  = 2 * hs + ph;
  const int ws0 = wave * 16;

  const int hi   = lane >> 4;      // half-wave select
  const int mrow = lane & 15;      // A-matrix row (pixel) this lane loads
  const int ncol = lane & 15;      // B/D column (Cout) this lane holds

  const int ndi = (ph == 0) ? 3 : 2;   // i in {ph, ph+2, ...}
  const int ndj = (pw == 0) ? 3 : 2;   // j in {pw, pw+2, ...}
  const int ntaps = ndi * ndj;

  // ---- stage B fragments for the taps of this parity class into LDS ----
  // slot s -> ti = s/ndj, tj = s%ndj -> tap = (ph+2*ti)*5 + (pw+2*tj)
  for (int idx = threadIdx.x; idx < ntaps * (TAP_HALVES / 8); idx += 256) {
    int s  = idx / (TAP_HALVES / 8);
    int ch = idx - s * (TAP_HALVES / 8);
    int ti = s / ndj, tj = s - ti * ndj;
    int tap = (ph + 2 * ti) * KK + (pw + 2 * tj);
    *(v8h*)(lds_w + (size_t)s * TAP_HALVES + ch * 8) =
        *(const v8h*)(wfrag + (size_t)tap * TAP_HALVES + ch * 8);
  }
  __syncthreads();

  // accumulators initialized with bias
  v8f acc[4];
#pragma unroll
  for (int nt = 0; nt < 4; ++nt) {
    float bv = bias[nt * 16 + ncol];
#pragma unroll
    for (int r = 0; r < 8; ++r) acc[nt][r] = bv;
  }

  int slot = 0;
  for (int i = ph; i < KK; i += 2) {
    const int di = (ph + i - 2) / 2;
    const _Float16* arowbase =
        xdpad + (size_t)(b * XDP + hs + di + 1) * XDP * CIN;
    for (int j = pw; j < KK; j += 2, ++slot) {
      const int dj = (pw + j - 2) / 2;
      const int tap = i * KK + j;
      // A fragment source: 16-bit A layout -> lane(mrow) reads two 16B chunks per k-step:
      //   chunk0: K = ks*32 + hi*8 + [0..7], chunk1: K = ks*32 + 16 + hi*8 + [0..7]
      const _Float16* aptr =
          arowbase + (size_t)(ws0 + mrow + dj + 1) * CIN + hi * 8;
      if (j + 2 < KK) {
        __builtin_prefetch(arowbase + (size_t)(ws0 + mrow + dj + 2) * CIN, 0, 0);
      }

      // per-pixel kern scales for this lane's 8 D rows (pixels hi*8 + r)
      const float* kp = kern2 + (((size_t)(b * 25 + tap) * TGT + h) * TGT)
                      + pw * HSUB + ws0 + hi * 8;
      v4f kv0 = *(const v4f*)kp;
      v4f kv1 = *(const v4f*)(kp + 4);

      v8f y[4] = {};
#pragma unroll
      for (int ks = 0; ks < 4; ++ks) {
        union { v16h v; v8h h8[2]; } a;
        a.h8[0] = *(const v8h*)(aptr + ks * 32);
        a.h8[1] = *(const v8h*)(aptr + ks * 32 + 16);
#pragma unroll
        for (int nt = 0; nt < 4; ++nt) {
          union { v16h v; v8h h8[2]; } bm;
          const _Float16* bp2 =
              lds_w + (((size_t)slot * 16 + ks * 4 + nt) * 32 + lane) * 16;
          bm.h8[0] = *(const v8h*)bp2;
          bm.h8[1] = *(const v8h*)(bp2 + 8);
          y[nt] = __builtin_amdgcn_wmma_f32_16x16x32_f16(
              /*neg_a=*/false, a.v, /*neg_b=*/false, bm.v,
              /*c_mod=*/(short)0, y[nt], /*reuse_a=*/false, /*reuse_b=*/false);
        }
      }
#pragma unroll
      for (int nt = 0; nt < 4; ++nt) {
#pragma unroll
        for (int r = 0; r < 4; ++r) acc[nt][r]     += kv0[r] * y[nt][r];
#pragma unroll
        for (int r = 0; r < 4; ++r) acc[nt][4 + r] += kv1[r] * y[nt][4 + r];
      }
    }
  }

  // store: lane holds Cout = nt*16 + ncol, pixels hi*8 + r  ->  w = 2*(ws0+pixel)+pw (stride 2)
  const int wbase = 2 * (ws0 + hi * 8) + pw;
#pragma unroll
  for (int nt = 0; nt < 4; ++nt) {
    float* op = out + (((size_t)b * COUT + nt * 16 + ncol) * TGT + h) * TGT + wbase;
#pragma unroll
    for (int r = 0; r < 8; ++r) op[2 * r] = acc[nt][r];
  }
}

extern "C" void kernel_launch(void* const* d_in, const int* in_sizes, int n_in,
                              void* d_out, int out_size, void* d_ws, size_t ws_size,
                              hipStream_t stream) {
  const float* x      = (const float*)d_in[0];
  const float* guide  = (const float*)d_in[1];
  const float* weight = (const float*)d_in[2];
  const float* bias   = (const float*)d_in[3];
  // d_in[4] = target_size (constant 256 for this problem)

  char* ws = (char*)d_ws;
  _Float16* xdpad = (_Float16*)ws;
  float*    kern2 = (float*)(ws + XDPAD_BYTES);
  _Float16* wfrag = (_Float16*)(ws + XDPAD_BYTES + KERN_BYTES);
  float*    outp  = (float*)d_out;

  // zero border of padded activations (interior overwritten by k_downsample)
  hipMemsetAsync(xdpad, 0, XDPAD_BYTES, stream);

  {
    int n = B_ * CIN * HSUB * HSUB;                 // 16,777,216
    k_downsample<<<n / 256, 256, 0, stream>>>(x, xdpad);
  }
  {
    int n = B_ * 25 * TGT * TGT;                    // 13,107,200
    k_pack<<<n / 256, 256, 0, stream>>>(guide, kern2);
  }
  {
    int n = (int)WFRAG_ELEMS;                       // 204,800
    k_wfrag<<<n / 256, 256, 0, stream>>>(weight, wfrag);
  }
  {
    int nblocks = B_ * 4 * HSUB;                    // 4096
    k_pac_wmma<<<nblocks, 256, LDS_BYTES, stream>>>(xdpad, kern2, wfrag, bias, outp);
  }
}
